// GeoAttention_37048387895861
// MI455X (gfx1250) — matrix operand order repaired and checked by reading, not compiled, not run on hardware
//
#include <hip/hip_runtime.h>
#include <hip/hip_bf16.h>

// ---------------------------------------------------------------------------
// Types for WMMA fragments (gfx1250, wave32)
// ---------------------------------------------------------------------------
typedef __attribute__((ext_vector_type(16))) __bf16 v16bf;
typedef __attribute__((ext_vector_type(8)))  __bf16 v8bf;
typedef __attribute__((ext_vector_type(8)))  float  v8f;
typedef unsigned int u32x4 __attribute__((ext_vector_type(4)));
typedef int          i32x8 __attribute__((ext_vector_type(8)));
typedef int          i32x4 __attribute__((ext_vector_type(4)));

#define BHD   4
#define NHEAD 12
#define NCTX  2048
#define CDIM  768
#define HDIM  64
#define QKSCALE 0.125f  // 64^-0.5

__device__ __forceinline__ __bf16 f2bf(float f) { return (__bf16)f; }

__device__ __forceinline__ v16bf cat16(v8bf lo, v8bf hi) {
    return __builtin_shufflevector(lo, hi, 0,1,2,3,4,5,6,7,8,9,10,11,12,13,14,15);
}

__device__ __forceinline__ v8f wmma_bf16(v16bf a, v16bf b, v8f c) {
    return __builtin_amdgcn_wmma_f32_16x16x32_bf16(
        /*neg_a=*/false, a, /*neg_b=*/false, b,
        /*c_mod=*/(short)0, c, /*reuse_a=*/false, /*reuse_b=*/false);
}

// CDNA5 async copy: 16B global -> LDS per lane, tracked by ASYNCcnt.
// LDS address = low 32 bits of the generic pointer (aperture scheme).
__device__ __forceinline__ void async_copy_b128(void* lds, const void* gaddr) {
    unsigned lo = (unsigned)(size_t)lds;
    asm volatile("global_load_async_to_lds_b128 %0, %1, off"
                 :: "v"(lo), "v"(gaddr) : "memory");
}
__device__ __forceinline__ void wait_async0() {
    asm volatile("s_wait_asynccnt 0" ::: "memory");
}

// ---------------------------------------------------------------------------
// Tensor Data Mover: 2D bf16 tile global->LDS, one instruction per tile.
// D# built per CDNA5 ISA 8.3/8.4 (group0: count/lds/global/type,
// group1: data_size=2B, dims, dim0-stride, LDS row padding).
// This toolchain's builtin takes 6 args (g0, g1, g2, g3, g4, cpol).
// ---------------------------------------------------------------------------
#if __has_builtin(__builtin_amdgcn_tensor_load_to_lds)
#define HAVE_TDM 1
__device__ __forceinline__ void tdm_load_tile_2d(
    unsigned lds_byte_addr, const void* gaddr,
    unsigned tensor_d0, unsigned tensor_d1, unsigned stride0_elems,
    unsigned tile_d0, unsigned tile_d1,
    unsigned pad_en, unsigned pad_interval_code, unsigned pad_amount_code) {
    unsigned long long ga = (unsigned long long)(size_t)gaddr;
    u32x4 g0;
    g0[0] = 1u;                                              // count=1
    g0[1] = lds_byte_addr;                                   // [63:32] lds_addr
    g0[2] = (unsigned)ga;                                    // global_addr lo
    g0[3] = ((unsigned)(ga >> 32) & 0x01FFFFFFu) | (2u << 30); // hi | type=2
    i32x8 g1;
    unsigned flags = (1u << 16)                              // data_size: 2 bytes
                   | (pad_en << 20) | (pad_interval_code << 22)
                   | (pad_amount_code << 25);
    g1[0] = (int)flags;                                      // mask=0 | flags
    g1[1] = (int)((tensor_d0 & 0xFFFFu) << 16);              // dim0[15:0] @ bit48
    g1[2] = (int)((tensor_d0 >> 16) | ((tensor_d1 & 0xFFFFu) << 16));
    g1[3] = (int)((tensor_d1 >> 16) | (tile_d0 << 16));      // tile_dim0 @ bit112
    g1[4] = (int)tile_d1;                                    // tile_dim1 (dim2=0)
    g1[5] = (int)stride0_elems;                              // dim0_stride lo32
    g1[6] = 0;
    g1[7] = 0;
    i32x4 z4 = {};
    i32x8 z8 = {};
    __builtin_amdgcn_tensor_load_to_lds(g0, g1, z4, z4, z8, 0);
}
#else
#define HAVE_TDM 0
#endif

// ---------------------------------------------------------------------------
// 16-lane butterfly max via v_permlane16_b32 (pure VALU, no LDS round trip).
// Selectors encode lane^1 / lane^2 / lane^4 / lane^8 permutations.
// ---------------------------------------------------------------------------
__device__ __forceinline__ float plmax(float v, unsigned s0, unsigned s1) {
#if __has_builtin(__builtin_amdgcn_permlane16)
    unsigned u = __float_as_uint(v);
    unsigned p = __builtin_amdgcn_permlane16(u, u, s0, s1, false, false);
    return fmaxf(v, __uint_as_float(p));
#else
    (void)s0; (void)s1;
    return v;
#endif
}
__device__ __forceinline__ float rowmax16(float v) {
#if __has_builtin(__builtin_amdgcn_permlane16)
    v = plmax(v, 0x67452301u, 0xEFCDAB89u);  // xor 1
    v = plmax(v, 0x54761032u, 0xDCFE98BAu);  // xor 2
    v = plmax(v, 0x32107654u, 0xBA98FEDCu);  // xor 4
    v = plmax(v, 0xFEDCBA98u, 0x76543210u);  // xor 8
#else
    #pragma unroll
    for (int sh = 1; sh < 16; sh <<= 1)
        v = fmaxf(v, __shfl_xor(v, sh, 32));
#endif
    return v;
}

// ---------------------------------------------------------------------------
// Kernel 1: fp32 -> bf16 conversion
// ---------------------------------------------------------------------------
__global__ __launch_bounds__(256) void cvt_f32_bf16(
    const float* __restrict__ src, __bf16* __restrict__ dst, int n) {
    int i = (blockIdx.x * 256 + threadIdx.x) * 4;
    if (i + 3 < n) {
        float4 f = *(const float4*)(src + i);
        dst[i + 0] = f2bf(f.x);
        dst[i + 1] = f2bf(f.y);
        dst[i + 2] = f2bf(f.z);
        dst[i + 3] = f2bf(f.w);
    } else {
        for (; i < n; ++i) dst[i] = f2bf(src[i]);
    }
}

// ---------------------------------------------------------------------------
// Kernel 2/4: bf16 GEMM  C[M,Nout] = A[M,K] @ B[K,Nout]
//   Block tile 128x128, 256 threads = 8 waves (2x4), wave tile 64x32.
//   A tile staged with global_load_async_to_lds_b128 (ASYNCcnt path);
//   B tile staged transposed through VGPRs.
// ---------------------------------------------------------------------------
template<int MODE>
__global__ __launch_bounds__(256) void gemm_bf16_wmma(
    const __bf16* __restrict__ A, const __bf16* __restrict__ Bw,
    int K, int Nout,
    __bf16* __restrict__ Qo, __bf16* __restrict__ Ko, __bf16* __restrict__ Vo,
    float* __restrict__ outF, const float* __restrict__ bias) {

    constexpr int BM = 128, BN = 128, BK = 32, LDA = BK + 8;
    __shared__ __align__(16) __bf16 As[BM][LDA];   // [m][k]
    __shared__ __align__(16) __bf16 Bs[BN][LDA];   // transposed: [n][k]

    const int tid  = threadIdx.x;
    const int wave = tid >> 5, lane = tid & 31;
    const int g = lane >> 4, ln = lane & 15;
    const int wm = wave >> 2, wn = wave & 3;
    const int m0 = blockIdx.x * BM, n0 = blockIdx.y * BN;

    v8f acc[4][2] = {};

    for (int k0 = 0; k0 < K; k0 += BK) {
        {   // A tile via async global->LDS
            int row = tid >> 1, half = tid & 1;
            const __bf16* gp = A + (size_t)(m0 + row) * K + k0 + 16 * half;
            async_copy_b128(&As[row][16 * half], gp);
            async_copy_b128(&As[row][16 * half + 8], gp + 8);
        }
        {   // B tile transposed via VGPRs
            int kk = tid >> 3, nseg = tid & 7;
            const __bf16* gp = Bw + (size_t)(k0 + kk) * Nout + n0 + 16 * nseg;
            v8bf w0 = *(const v8bf*)gp;
            v8bf w1 = *(const v8bf*)(gp + 8);
            #pragma unroll
            for (int e = 0; e < 8; ++e) {
                Bs[16 * nseg + e][kk]     = w0[e];
                Bs[16 * nseg + 8 + e][kk] = w1[e];
            }
        }
        if (k0 + BK < K)
            __builtin_prefetch(Bw + (size_t)(k0 + BK + (tid >> 3)) * Nout + n0, 0, 0);
        wait_async0();
        __syncthreads();

        #pragma unroll
        for (int mt = 0; mt < 4; ++mt) {
            const __bf16* ap = &As[wm * 64 + mt * 16 + ln][0];
            v16bf af = cat16(*(const v8bf*)(ap + 8 * g),
                             *(const v8bf*)(ap + 16 + 8 * g));
            #pragma unroll
            for (int nt = 0; nt < 2; ++nt) {
                const __bf16* bp = &Bs[wn * 32 + nt * 16 + ln][0];
                v16bf bf_ = cat16(*(const v8bf*)(bp + 8 * g),
                                  *(const v8bf*)(bp + 16 + 8 * g));
                acc[mt][nt] = wmma_bf16(af, bf_, acc[mt][nt]);
            }
        }
        __syncthreads();
    }

    #pragma unroll
    for (int mt = 0; mt < 4; ++mt) {
        #pragma unroll
        for (int nt = 0; nt < 2; ++nt) {
            const int col = n0 + wn * 32 + nt * 16 + ln;
            #pragma unroll
            for (int r = 0; r < 8; ++r) {
                const int row = m0 + wm * 64 + mt * 16 + 8 * g + r;
                float val = acc[mt][nt][r];
                if (MODE == 0) {
                    const int t = col / CDIM;
                    const int h = (col % CDIM) >> 6;
                    const int d = col & 63;
                    const int b = row >> 11, n = row & (NCTX - 1);
                    const size_t idx =
                        (((size_t)b * NHEAD + h) * NCTX + n) * HDIM + d;
                    if (t == 0)      Qo[idx] = f2bf(val * QKSCALE);
                    else if (t == 1) Ko[idx] = f2bf(val);
                    else             Vo[idx] = f2bf(val);
                } else {
                    outF[(size_t)row * Nout + col] = val + bias[col];
                }
            }
        }
    }
}

// ---------------------------------------------------------------------------
// Kernel 3: flash attention. K tile staged by the Tensor Data Mover (one
// tensor_load_to_lds per key block, TENSORcnt); V transposed via VGPRs;
// row-sum on the matrix engine (P @ ones); row-max via v_permlane16_b32.
// ---------------------------------------------------------------------------
__global__ __launch_bounds__(256) void attn_flash_wmma(
    const __bf16* __restrict__ Q, const __bf16* __restrict__ K,
    const __bf16* __restrict__ V, __bf16* __restrict__ O) {

    constexpr int KB = 64, LDK = HDIM + 8;
    __shared__ __align__(16) __bf16 Ks[KB][LDK];        // [key][d]
    __shared__ __align__(16) __bf16 Vt[HDIM][LDK];      // [d][key]
    __shared__ __align__(16) __bf16 Ps[8][16][LDK];     // per-wave P tile

    const int tid  = threadIdx.x;
    const int wave = tid >> 5, lane = tid & 31;
    const int g = lane >> 4, ln = lane & 15;
    const int bh = blockIdx.y;
    const int q0 = blockIdx.x * 128;
    const size_t base = (size_t)bh * NCTX * HDIM;

    v16bf onesB;
    #pragma unroll
    for (int e = 0; e < 16; ++e) onesB[e] = (__bf16)1.0f;

    const __bf16* qrow = Q + base + (size_t)(q0 + wave * 16 + ln) * HDIM;
    v16bf qa[2];
    #pragma unroll
    for (int f = 0; f < 2; ++f)
        qa[f] = cat16(*(const v8bf*)(qrow + 32 * f + 8 * g),
                      *(const v8bf*)(qrow + 32 * f + 16 + 8 * g));

    v8f o[4] = {};
    v8f rsum = {};
    float rmax[8];
    #pragma unroll
    for (int r = 0; r < 8; ++r) rmax[r] = -3.0e38f;
    const float LOG2E = 1.4426950408889634f;

    for (int kb = 0; kb < NCTX; kb += KB) {
        // ---- stage K tile ----
#if HAVE_TDM
        if (wave == 0) {
            // 64x64 bf16 tile; LDS rows padded: 32-DWORD interval (code 4),
            // 4-DWORD pad (code 3) -> Ks[64][72] layout produced by the DMA.
            tdm_load_tile_2d((unsigned)(size_t)&Ks[0][0],
                             K + base + (size_t)kb * HDIM,
                             /*tensor_d0=*/HDIM, /*tensor_d1=*/KB,
                             /*stride0=*/HDIM,
                             /*tile_d0=*/HDIM, /*tile_d1=*/KB,
                             /*pad_en=*/1u, /*interval=*/4u, /*amount=*/3u);
        }
#else
        {
            int row = tid >> 2, seg = tid & 3;
            const __bf16* gk = K + base + (size_t)(kb + row) * HDIM + 16 * seg;
            async_copy_b128(&Ks[row][16 * seg], gk);
            async_copy_b128(&Ks[row][16 * seg + 8], gk + 8);
        }
#endif
        // ---- stage V transposed via VGPRs ----
        {
            int row = tid >> 2, seg = tid & 3;
            const __bf16* gv = V + base + (size_t)(kb + row) * HDIM + 16 * seg;
            v8bf v0 = *(const v8bf*)gv;
            v8bf v1 = *(const v8bf*)(gv + 8);
            #pragma unroll
            for (int e = 0; e < 8; ++e) {
                Vt[16 * seg + e][row]     = v0[e];
                Vt[16 * seg + 8 + e][row] = v1[e];
            }
        }
#if HAVE_TDM
        __builtin_amdgcn_s_wait_tensorcnt(0);
#else
        wait_async0();
#endif
        __syncthreads();

        // ---- S = Q @ K^T ----
        v8f st[4];
        #pragma unroll
        for (int kt = 0; kt < 4; ++kt) {
            v8f s = {};
            const __bf16* bp = &Ks[16 * kt + ln][0];
            #pragma unroll
            for (int ds = 0; ds < 2; ++ds) {
                v16bf bf_ = cat16(*(const v8bf*)(bp + 32 * ds + 8 * g),
                                  *(const v8bf*)(bp + 32 * ds + 16 + 8 * g));
                s = wmma_bf16(qa[ds], bf_, s);
            }
            st[kt] = s;
        }

        // ---- online softmax: permlane row-max, P -> LDS, rescale state ----
        #pragma unroll
        for (int r = 0; r < 8; ++r) {
            float mr = rowmax16(fmaxf(fmaxf(st[0][r], st[1][r]),
                                      fmaxf(st[2][r], st[3][r])));
            float mnew = fmaxf(rmax[r], mr);
            float alpha = exp2f((rmax[r] - mnew) * LOG2E);
            rmax[r] = mnew;
            rsum[r] *= alpha;
            #pragma unroll
            for (int dt = 0; dt < 4; ++dt) o[dt][r] *= alpha;
            #pragma unroll
            for (int kt = 0; kt < 4; ++kt) {
                float p = exp2f((st[kt][r] - mnew) * LOG2E);
                Ps[wave][8 * g + r][16 * kt + ln] = f2bf(p);
            }
        }
        asm volatile("s_wait_dscnt 0" ::: "memory");     // same-wave LDS RAW

        // ---- P A-fragments ----
        v16bf pa[2];
        const __bf16* prow = &Ps[wave][ln][0];
        #pragma unroll
        for (int f = 0; f < 2; ++f)
            pa[f] = cat16(*(const v8bf*)(prow + 32 * f + 8 * g),
                          *(const v8bf*)(prow + 32 * f + 16 + 8 * g));

        // ---- row-sum on the matrix engine, then O += P @ V ----
        rsum = wmma_bf16(pa[0], onesB, rsum);
        rsum = wmma_bf16(pa[1], onesB, rsum);
        #pragma unroll
        for (int dt = 0; dt < 4; ++dt) {
            const __bf16* vp = &Vt[16 * dt + ln][0];
            #pragma unroll
            for (int ks = 0; ks < 2; ++ks) {
                v16bf bf_ = cat16(*(const v8bf*)(vp + 32 * ks + 8 * g),
                                  *(const v8bf*)(vp + 32 * ks + 16 + 8 * g));
                o[dt] = wmma_bf16(pa[ks], bf_, o[dt]);
            }
        }
        __syncthreads();
    }

    // ---- finalize: O[b][n][h*64+d] = o / rsum ----
    const int b = bh / NHEAD, h = bh % NHEAD;
    #pragma unroll
    for (int r = 0; r < 8; ++r) {
        float inv = 1.0f / rsum[r];
        int n = q0 + wave * 16 + 8 * g + r;
        #pragma unroll
        for (int dt = 0; dt < 4; ++dt) {
            size_t idx = ((size_t)b * NCTX + n) * CDIM + h * HDIM + 16 * dt + ln;
            O[idx] = f2bf(o[dt][r] * inv);
        }
    }
}

// ---------------------------------------------------------------------------
// Host launcher
// ---------------------------------------------------------------------------
extern "C" void kernel_launch(void* const* d_in, const int* in_sizes, int n_in,
                              void* d_out, int out_size, void* d_ws, size_t ws_size,
                              hipStream_t stream) {
    const float* x     = (const float*)d_in[0];
    const float* Wqkv  = (const float*)d_in[1];
    const float* Wproj = (const float*)d_in[2];
    const float* bproj = (const float*)d_in[3];
    float* out = (float*)d_out;

    const size_t M = (size_t)BHD * NCTX;
    const size_t SZ_XB    = M * CDIM;
    const size_t SZ_WQKV  = (size_t)CDIM * 3 * CDIM;
    const size_t SZ_WPROJ = (size_t)CDIM * CDIM;
    const size_t SZ_HEADS = (size_t)BHD * NHEAD * NCTX * HDIM;

    unsigned char* w = (unsigned char*)d_ws;
    __bf16* xb    = (__bf16*)w;  w += SZ_XB    * 2;
    __bf16* wqkvb = (__bf16*)w;  w += SZ_WQKV  * 2;
    __bf16* wprjb = (__bf16*)w;  w += SZ_WPROJ * 2;
    __bf16* Qb    = (__bf16*)w;  w += SZ_HEADS * 2;
    __bf16* Kb    = (__bf16*)w;  w += SZ_HEADS * 2;
    __bf16* Vb    = (__bf16*)w;  w += SZ_HEADS * 2;
    __bf16* Ob    = (__bf16*)w;  w += SZ_XB    * 2;

    cvt_f32_bf16<<<(unsigned)((SZ_XB / 4 + 255) / 256), 256, 0, stream>>>(x, xb, (int)SZ_XB);
    cvt_f32_bf16<<<(unsigned)((SZ_WQKV / 4 + 255) / 256), 256, 0, stream>>>(Wqkv, wqkvb, (int)SZ_WQKV);
    cvt_f32_bf16<<<(unsigned)((SZ_WPROJ / 4 + 255) / 256), 256, 0, stream>>>(Wproj, wprjb, (int)SZ_WPROJ);

    {
        dim3 grid((unsigned)(M / 128), (3 * CDIM) / 128);
        gemm_bf16_wmma<0><<<grid, 256, 0, stream>>>(
            xb, wqkvb, CDIM, 3 * CDIM, Qb, Kb, Vb, nullptr, nullptr);
    }
    {
        dim3 grid(NCTX / 128, BHD * NHEAD);
        attn_flash_wmma<<<grid, 256, 0, stream>>>(Qb, Kb, Vb, Ob);
    }
    {
        dim3 grid((unsigned)(M / 128), CDIM / 128);
        gemm_bf16_wmma<1><<<grid, 256, 0, stream>>>(
            Ob, wprjb, CDIM, CDIM, nullptr, nullptr, nullptr, out, bproj);
    }
}
